// Transformer_76802605187779
// MI455X (gfx1250) — matrix operand order repaired and checked by reading, not compiled, not run on hardware
//
#include <hip/hip_runtime.h>
#include <hip/hip_bf16.h>

// ---------------------------------------------------------------------------
// Transformer forward for gfx1250 (MI455X, wave32, WMMA 16x16x32 f16).
// ---------------------------------------------------------------------------

typedef __attribute__((ext_vector_type(16))) _Float16 v16h;
typedef __attribute__((ext_vector_type(8)))  _Float16 v8h;
typedef __attribute__((ext_vector_type(8)))  float    v8f;

#define Bk  2
#define Sk  2048
#define Dk  512
#define Hk  8
#define Lk  4
#define Fk  2048
#define Vk  32000
#define DHk 64
#define Mk  (Bk*Sk)   // 4096 tokens

union V16 { v16h v; v8h h[2]; };
union V8F { v8f v; float f[8]; };

__device__ __forceinline__ v8f wmma_f16(v16h a, v16h b, v8f c) {
  // D = A(16x32 f16) x B(32x16 f16) + C(16x16 f32)
  return __builtin_amdgcn_wmma_f32_16x16x32_f16(
      /*neg_a=*/false, a, /*neg_b=*/false, b,
      /*c_mod=*/(short)0, c, /*reuse_a=*/false, /*reuse_b=*/false);
}

// ---------------------------------------------------------------------------
// h[b,s,:] = emb[x[b,s]] + sinusoidal positional encoding
// ---------------------------------------------------------------------------
__global__ void embed_kernel(const int* __restrict__ x, const float* __restrict__ emb,
                             float* __restrict__ h) {
  size_t i = (size_t)blockIdx.x * blockDim.x + threadIdx.x;   // over Mk*Dk
  int d = (int)(i % Dk);
  size_t t = i / Dk;
  int s = (int)(t % Sk);
  int tok = x[t];
  int p = d >> 1;
  const float c = 9.210340371976184f / (float)Dk;             // ln(10000)/D
  float ang = (float)s * __expf(-(float)(2 * p) * c);
  float pe = (d & 1) ? __cosf(ang) : __sinf(ang);
  h[i] = emb[(size_t)tok * Dk + d] + pe;
}

// ---------------------------------------------------------------------------
// LayerNorm over D=512, one wave per row, f16 output (feeds WMMA A-side)
// ---------------------------------------------------------------------------
__global__ void ln_f16_kernel(const float* __restrict__ h, const float* __restrict__ g,
                              const float* __restrict__ b, _Float16* __restrict__ out) {
  int lane = threadIdx.x & 31;
  size_t row = (size_t)blockIdx.x * 8 + (threadIdx.x >> 5);
  const float* xp = h + row * Dk;
  float s = 0.f, ss = 0.f;
  for (int i = lane; i < Dk; i += 32) { float v = xp[i]; s += v; ss += v * v; }
  for (int o = 16; o > 0; o >>= 1) { s += __shfl_xor(s, o, 32); ss += __shfl_xor(ss, o, 32); }
  float mean = s * (1.f / Dk);
  float var  = ss * (1.f / Dk) - mean * mean;
  float inv  = rsqrtf(var + 1e-5f);
  _Float16* yp = out + row * Dk;
  for (int i = lane; i < Dk; i += 32)
    yp[i] = (_Float16)((xp[i] - mean) * inv * g[i] + b[i]);
}

// ---------------------------------------------------------------------------
// Pack f32 weight [K,N] into fragment-contiguous f16 tiles.
// Tile (kt, nt) holds a 32x16 B-fragment; within it lane L owns 16 halves:
//   half j -> element B[kt*32 + 16*(L>=16) + j][nt*16 + (L&15)]
// so the GEMM loads one contiguous 32B v16h per lane per fragment.
// ---------------------------------------------------------------------------
__global__ void pack_w_kernel(const float* __restrict__ W, _Float16* __restrict__ P,
                              int K, int N) {
  size_t i = (size_t)blockIdx.x * blockDim.x + threadIdx.x;
  if (i >= (size_t)K * N) return;
  int j = (int)(i & 15);
  size_t t = i >> 4;
  int lane = (int)(t & 31); t >>= 5;
  int nt = (int)(t % (size_t)(N >> 4));
  int kt = (int)(t / (size_t)(N >> 4));
  int k = kt * 32 + ((lane >> 4) ? 16 : 0) + j;
  int n = nt * 16 + (lane & 15);
  P[i] = (_Float16)W[(size_t)k * N + n];
}

// ---------------------------------------------------------------------------
// GEMM: out[M,N] = A[M,K](f16) x Bp(packed f16) + bias, optional residual/relu,
// f32 or f16 store. One wave computes a 32x64 output tile: two A fragments
// reuse the same four B fragments (8 WMMA / k-step), halving B-side traffic.
// B stream prefetched two k-steps ahead (global_prefetch_b8).
// ---------------------------------------------------------------------------
__global__ void gemm_wmma_kernel(const _Float16* __restrict__ A, const _Float16* __restrict__ Bp,
                                 const float* __restrict__ bias, const float* __restrict__ residual,
                                 float* __restrict__ outF, _Float16* __restrict__ outH,
                                 int M, int N, int K, int relu) {
  int lane  = threadIdx.x & 31;
  int lhalf = lane >> 4, lmod = lane & 15;
  int wid = blockIdx.x * (blockDim.x >> 5) + (threadIdx.x >> 5);
  int tilesN = N >> 6;
  int tm = wid / tilesN, tn = wid % tilesN;
  if (tm * 32 >= M) return;

  v8f acc[2][4] = {};
  const _Float16* arow0 = A + (size_t)(tm * 32 + lmod) * K + lhalf * 16;
  const _Float16* arow1 = arow0 + (size_t)16 * K;
  const _Float16* bbase = Bp + ((size_t)tn * 4) * 512 + lane * 16;
  size_t bstepK = (size_t)(N >> 4) * 512;

  for (int k0 = 0; k0 < K; k0 += 32) {
    V16 a0, a1;
    a0.h[0] = *(const v8h*)(arow0 + k0);        // K = k0 + base + 0..7
    a0.h[1] = *(const v8h*)(arow0 + k0 + 8);    // K = k0 + base + 8..15
    a1.h[0] = *(const v8h*)(arow1 + k0);
    a1.h[1] = *(const v8h*)(arow1 + k0 + 8);
    const _Float16* bp = bbase + (size_t)(k0 >> 5) * bstepK;
    __builtin_prefetch(bp + 2 * bstepK, 0, 1);  // global_prefetch_b8, 2 steps ahead
    v16h b0 = *(const v16h*)(bp);
    v16h b1 = *(const v16h*)(bp + 512);
    v16h b2 = *(const v16h*)(bp + 1024);
    v16h b3 = *(const v16h*)(bp + 1536);
    acc[0][0] = wmma_f16(a0.v, b0, acc[0][0]);
    acc[0][1] = wmma_f16(a0.v, b1, acc[0][1]);
    acc[0][2] = wmma_f16(a0.v, b2, acc[0][2]);
    acc[0][3] = wmma_f16(a0.v, b3, acc[0][3]);
    acc[1][0] = wmma_f16(a1.v, b0, acc[1][0]);
    acc[1][1] = wmma_f16(a1.v, b1, acc[1][1]);
    acc[1][2] = wmma_f16(a1.v, b2, acc[1][2]);
    acc[1][3] = wmma_f16(a1.v, b3, acc[1][3]);
  }

  #pragma unroll
  for (int rg = 0; rg < 2; ++rg) {
    #pragma unroll
    for (int sub = 0; sub < 4; ++sub) {
      V8F u; u.v = acc[rg][sub];
      int col = tn * 64 + sub * 16 + lmod;
      float bb = bias[col];
      #pragma unroll
      for (int r = 0; r < 8; ++r) {
        int row = tm * 32 + rg * 16 + r + lhalf * 8;
        float val = u.f[r] + bb;
        if (residual) val += residual[(size_t)row * N + col];
        if (relu)     val = fmaxf(val, 0.f);
        if (outF) outF[(size_t)row * N + col] = val;
        else      outH[(size_t)row * N + col] = (_Float16)val;
      }
    }
  }
}

// ---------------------------------------------------------------------------
// V [b,s,h*64+d] -> vT [b,h,d,s] so p@v B-fragments are contiguous per lane.
// ---------------------------------------------------------------------------
__global__ void transpose_v_kernel(const _Float16* __restrict__ v, _Float16* __restrict__ vT) {
  size_t i = (size_t)blockIdx.x * blockDim.x + threadIdx.x;   // over Mk*Dk
  int d = (int)(i % Dk);
  size_t t = i / Dk;
  int s = (int)(t % Sk);
  int b = (int)(t / Sk);
  int hh = d >> 6, dd = d & 63;
  vT[(((size_t)b * Hk + hh) * DHk + dd) * Sk + s] = v[i];
}

__global__ void cast_f16_kernel(const float* __restrict__ a, _Float16* __restrict__ o) {
  size_t i = (size_t)blockIdx.x * blockDim.x + threadIdx.x;
  o[i] = (_Float16)a[i];
}

// ---------------------------------------------------------------------------
// Streaming (flash-style) causal attention. One wave per 16-row query tile;
// kv chunks of 32; online softmax; q@k^T and p@v in WMMA; P converted from
// C-layout to A-layout through per-wave LDS (s_wait_dscnt between).
// ---------------------------------------------------------------------------
__global__ void attn_kernel(const _Float16* __restrict__ q, const _Float16* __restrict__ k,
                            const _Float16* __restrict__ vT, _Float16* __restrict__ o) {
  __shared__ _Float16 lds[8][16 * 40];     // per-wave 16x32 P tile, padded rows
  int w = threadIdx.x >> 5;
  int lane = threadIdx.x & 31;
  int lhalf = lane >> 4, lmod = lane & 15;
  int gw = blockIdx.x * 8 + w;
  int qt = gw & (Sk / 16 - 1);
  int bh = gw / (Sk / 16);
  int hh = bh & (Hk - 1);
  int bb = bh / Hk;
  const float scale = 0.125f;              // 1/sqrt(64)

  V16 qa[2];
  {
    const _Float16* qp = q + ((size_t)bb * Sk + qt * 16 + lmod) * Dk + hh * DHk + lhalf * 16;
    #pragma unroll
    for (int c = 0; c < 2; ++c) {
      qa[c].h[0] = *(const v8h*)(qp + c * 32);
      qa[c].h[1] = *(const v8h*)(qp + c * 32 + 8);
    }
  }

  float mrow[8], lrow[8], ofl[4][8];
  #pragma unroll
  for (int r = 0; r < 8; ++r) { mrow[r] = -1e30f; lrow[r] = 0.f; }
  #pragma unroll
  for (int c = 0; c < 4; ++c)
    #pragma unroll
    for (int r = 0; r < 8; ++r) ofl[c][r] = 0.f;

  int kvend = (qt + 1) * 16;
  for (int kv0 = 0; kv0 < kvend; kv0 += 32) {
    // ---- scores: S(16x32) = Q(16x64) @ K^T, two 16x16 tiles, K-dim 64 = 2 wmma
    v8f sc0 = {}, sc1 = {};
    #pragma unroll
    for (int c = 0; c < 2; ++c) {
      const _Float16* kp0 = k + ((size_t)bb * Sk + kv0 + lmod) * Dk + hh * DHk + c * 32 + lhalf * 16;
      v16h kb0 = *(const v16h*)kp0;
      v16h kb1 = *(const v16h*)(kp0 + (size_t)16 * Dk);
      sc0 = wmma_f16(qa[c].v, kb0, sc0);
      sc1 = wmma_f16(qa[c].v, kb1, sc1);
    }
    V8F s0, s1; s0.v = sc0; s1.v = sc1;
    bool domask = (kv0 + 31) > qt * 16;

    // ---- online softmax (row stats reduced across 16-lane groups)
    #pragma unroll
    for (int r = 0; r < 8; ++r) {
      int qrow = qt * 16 + r + lhalf * 8;
      float a0 = s0.f[r] * scale, a1 = s1.f[r] * scale;
      if (domask) {
        if (kv0 + lmod      > qrow) a0 = -1e30f;
        if (kv0 + 16 + lmod > qrow) a1 = -1e30f;
      }
      float loc = fmaxf(a0, a1);
      loc = fmaxf(loc, __shfl_xor(loc, 1, 32));
      loc = fmaxf(loc, __shfl_xor(loc, 2, 32));
      loc = fmaxf(loc, __shfl_xor(loc, 4, 32));
      loc = fmaxf(loc, __shfl_xor(loc, 8, 32));
      float nm = fmaxf(mrow[r], loc);
      float alpha = __expf(mrow[r] - nm);
      mrow[r] = nm;
      float e0 = __expf(a0 - nm), e1 = __expf(a1 - nm);
      float rs = e0 + e1;
      rs += __shfl_xor(rs, 1, 32);
      rs += __shfl_xor(rs, 2, 32);
      rs += __shfl_xor(rs, 4, 32);
      rs += __shfl_xor(rs, 8, 32);
      lrow[r] = lrow[r] * alpha + rs;
      #pragma unroll
      for (int c = 0; c < 4; ++c) ofl[c][r] *= alpha;
      int prow = r + lhalf * 8;
      lds[w][prow * 40 + lmod]      = (_Float16)e0;
      lds[w][prow * 40 + 16 + lmod] = (_Float16)e1;
    }
    asm volatile("s_wait_dscnt 0" ::: "memory");

    // ---- P tile reloaded in A-fragment layout
    V16 pa;
    {
      const _Float16* pl = &lds[w][lmod * 40 + lhalf * 16];
      pa.h[0] = *(const v8h*)pl;
      pa.h[1] = *(const v8h*)(pl + 8);
    }

    // ---- O(16x64) += P(16x32) @ V(32x64), 4 wmma
    #pragma unroll
    for (int c = 0; c < 4; ++c) {
      const _Float16* vp = vT + (((size_t)bb * Hk + hh) * DHk + c * 16 + lmod) * Sk + kv0 + lhalf * 16;
      v16h vb = *(const v16h*)vp;
      V8F oc;
      #pragma unroll
      for (int r = 0; r < 8; ++r) oc.f[r] = ofl[c][r];
      oc.v = wmma_f16(pa.v, vb, oc.v);
      #pragma unroll
      for (int r = 0; r < 8; ++r) ofl[c][r] = oc.f[r];
    }
  }

  #pragma unroll
  for (int c = 0; c < 4; ++c) {
    #pragma unroll
    for (int r = 0; r < 8; ++r) {
      int row = qt * 16 + r + lhalf * 8;
      float inv = 1.f / lrow[r];
      o[((size_t)bb * Sk + row) * Dk + hh * DHk + c * 16 + lmod] = (_Float16)(ofl[c][r] * inv);
    }
  }
}

// ---------------------------------------------------------------------------
// Host orchestration
// ---------------------------------------------------------------------------
extern "C" void kernel_launch(void* const* d_in, const int* in_sizes, int n_in,
                              void* d_out, int out_size, void* d_ws, size_t ws_size,
                              hipStream_t stream) {
  (void)in_sizes; (void)n_in; (void)out_size; (void)ws_size;
  const int*   x      = (const int*)  d_in[0];
  const float* emb    = (const float*)d_in[1];
  const float* attn_g = (const float*)d_in[2];
  const float* attn_b = (const float*)d_in[3];
  const float* wq     = (const float*)d_in[4];
  const float* bq     = (const float*)d_in[5];
  const float* wk     = (const float*)d_in[6];
  const float* bk     = (const float*)d_in[7];
  const float* wv     = (const float*)d_in[8];
  const float* bv     = (const float*)d_in[9];
  const float* wo     = (const float*)d_in[10];
  const float* bo     = (const float*)d_in[11];
  const float* mlp_g  = (const float*)d_in[12];
  const float* mlp_b  = (const float*)d_in[13];
  const float* w_up   = (const float*)d_in[14];
  const float* b_up   = (const float*)d_in[15];
  const float* w_down = (const float*)d_in[16];
  const float* b_down = (const float*)d_in[17];
  const float* w_un   = (const float*)d_in[18];
  const float* b_un   = (const float*)d_in[19];

  char* ws = (char*)d_ws;
  size_t off = 0;
  auto alloc = [&](size_t bytes) -> void* {
    void* p = ws + off;
    off = (off + bytes + 255) & ~(size_t)255;
    return p;
  };

  float*     h   = (float*)    alloc((size_t)Mk * Dk * 4);
  _Float16*  xn  = (_Float16*) alloc((size_t)Mk * Dk * 2);
  _Float16*  qb  = (_Float16*) alloc((size_t)Mk * Dk * 2);
  _Float16*  kb  = (_Float16*) alloc((size_t)Mk * Dk * 2);
  _Float16*  vb  = (_Float16*) alloc((size_t)Mk * Dk * 2);
  _Float16*  vT  = (_Float16*) alloc((size_t)Mk * Dk * 2);
  _Float16*  ob  = (_Float16*) alloc((size_t)Mk * Dk * 2);
  _Float16*  ub  = (_Float16*) alloc((size_t)Mk * Fk * 2);
  _Float16*  hf  = (_Float16*) alloc((size_t)Mk * Dk * 2);
  _Float16*  pwA = (_Float16*) alloc((size_t)Dk * Dk * 2);   // reused q/k/v/o
  _Float16*  pwU = (_Float16*) alloc((size_t)Dk * Fk * 2);
  _Float16*  pwD = (_Float16*) alloc((size_t)Fk * Dk * 2);
  _Float16*  pwV = (_Float16*) alloc((size_t)Dk * Vk * 2);

  const int TB = 256;
  auto pack = [&](const float* W, int K, int N, _Float16* P) {
    int grid = (int)(((size_t)K * N + TB - 1) / TB);
    pack_w_kernel<<<grid, TB, 0, stream>>>(W, P, K, N);
  };
  auto gemm = [&](const _Float16* A, const _Float16* Bp, const float* bias,
                  const float* residual, float* outF, _Float16* outH,
                  int M, int N, int K, int relu) {
    int tiles = (M / 32) * (N / 64);
    gemm_wmma_kernel<<<(tiles + 7) / 8, TB, 0, stream>>>(A, Bp, bias, residual, outF, outH,
                                                         M, N, K, relu);
  };

  // Embedding + positional encoding
  embed_kernel<<<(Mk * Dk) / TB, TB, 0, stream>>>(x, emb, h);

  for (int l = 0; l < Lk; ++l) {
    const size_t wdd = (size_t)l * Dk * Dk;
    // ---- attention block
    ln_f16_kernel<<<Mk / 8, TB, 0, stream>>>(h, attn_g + l * Dk, attn_b + l * Dk, xn);
    pack(wq + wdd, Dk, Dk, pwA);
    gemm(xn, pwA, bq + l * Dk, nullptr, nullptr, qb, Mk, Dk, Dk, 0);
    pack(wk + wdd, Dk, Dk, pwA);
    gemm(xn, pwA, bk + l * Dk, nullptr, nullptr, kb, Mk, Dk, Dk, 0);
    pack(wv + wdd, Dk, Dk, pwA);
    gemm(xn, pwA, bv + l * Dk, nullptr, nullptr, vb, Mk, Dk, Dk, 0);
    transpose_v_kernel<<<(Mk * Dk) / TB, TB, 0, stream>>>(vb, vT);
    attn_kernel<<<(Bk * Hk * (Sk / 16)) / 8, TB, 0, stream>>>(qb, kb, vT, ob);
    pack(wo + wdd, Dk, Dk, pwA);
    gemm(ob, pwA, bo + l * Dk, h, h, nullptr, Mk, Dk, Dk, 0);   // h += o @ Wo + bo
    // ---- MLP block
    ln_f16_kernel<<<Mk / 8, TB, 0, stream>>>(h, mlp_g + l * Dk, mlp_b + l * Dk, xn);
    pack(w_up + (size_t)l * Dk * Fk, Dk, Fk, pwU);
    gemm(xn, pwU, b_up + l * Fk, nullptr, nullptr, ub, Mk, Fk, Dk, 1 /*relu*/);
    pack(w_down + (size_t)l * Fk * Dk, Fk, Dk, pwD);
    gemm(ub, pwD, b_down + l * Dk, h, h, nullptr, Mk, Dk, Fk, 0); // h += relu(..) @ Wd + bd
  }

  // Unembedding: logits = h @ w_un + b_un
  cast_f16_kernel<<<(Mk * Dk) / TB, TB, 0, stream>>>(h, hf);
  pack(w_un, Dk, Vk, pwV);
  gemm(hf, pwV, b_un, nullptr, (float*)d_out, nullptr, Mk, Vk, Dk, 0);
}
